// MILossGaussian_25391846654578
// MI455X (gfx1250) — matrix operand order repaired and checked by reading, not compiled, not run on hardware
//
#include <hip/hip_runtime.h>

typedef __attribute__((ext_vector_type(16))) _Float16 v16h;
typedef __attribute__((ext_vector_type(8)))  float    v8f;

// exp(-2048*d^2) computed as 2^(C2*d^2), C2 = -2048*log2(e)
#define C2LOG    (-2954.6394437f)
#define BIN_STEP (1.0f / 31.0f)    // linspace(0,1,32) spacing
#define MI_EPS   1e-10f

// ---- monotone float<->uint encoding for order-independent integer atomics ----
__device__ __forceinline__ unsigned flipU(float f) {
  unsigned u = __float_as_uint(f);
  return u ^ ((unsigned)((int)u >> 31) | 0x80000000u);
}
__device__ __forceinline__ float unflipU(unsigned u) {
  unsigned v = (u & 0x80000000u) ? (u ^ 0x80000000u) : ~u;
  return __uint_as_float(v);
}

// 2^(quadratic): two v_fma_f32 + one v_exp_f32
__device__ __forceinline__ float gaussw(float x, float qb, float qc) {
  return __builtin_amdgcn_exp2f(__builtin_fmaf(__builtin_fmaf(C2LOG, x, qb), x, qc));
}

__global__ void mi_init_kernel(unsigned* mm) {
  int t = threadIdx.x;
  if (t < 16) mm[t] = (t & 1) ? 0u : 0xFFFFFFFFu;  // even = min slot, odd = max slot
}

__global__ void __launch_bounds__(256) mi_minmax_kernel(
    const float* __restrict__ tar, const float* __restrict__ src,
    unsigned* __restrict__ mm, int P) {
  __shared__ float sh[4][256];
  int sample = blockIdx.y;
  int tid = threadIdx.x;
  float tmn = 3.4e38f, tmx = -3.4e38f, smn = 3.4e38f, smx = -3.4e38f;
  const float* t = tar + (size_t)sample * P;
  const float* s = src + (size_t)sample * P;
  for (int i = blockIdx.x * blockDim.x + tid; i < P; i += gridDim.x * blockDim.x) {
    float tv = t[i], sv = s[i];
    tmn = fminf(tmn, tv); tmx = fmaxf(tmx, tv);
    smn = fminf(smn, sv); smx = fmaxf(smx, sv);
  }
  sh[0][tid] = tmn; sh[1][tid] = tmx; sh[2][tid] = smn; sh[3][tid] = smx;
  __syncthreads();
  for (int st = 128; st > 0; st >>= 1) {
    if (tid < st) {
      sh[0][tid] = fminf(sh[0][tid], sh[0][tid + st]);
      sh[1][tid] = fmaxf(sh[1][tid], sh[1][tid + st]);
      sh[2][tid] = fminf(sh[2][tid], sh[2][tid + st]);
      sh[3][tid] = fmaxf(sh[3][tid], sh[3][tid + st]);
    }
    __syncthreads();
  }
  if (tid == 0) {
    atomicMin(&mm[sample * 4 + 0], flipU(sh[0][0]));
    atomicMax(&mm[sample * 4 + 1], flipU(sh[1][0]));
    atomicMin(&mm[sample * 4 + 2], flipU(sh[2][0]));
    atomicMax(&mm[sample * 4 + 3], flipU(sh[3][0]));
  }
}

// Fused soft-binning + 32x32xK joint-histogram GEMM via v_wmma_f32_16x16x32_f16.
// One wave owns all four 16x16 tiles of the 32x32 accumulator; A/B fragments are
// generated in-register in a contiguous k order (k = 16*(lane/16) + element) shared by A and B.
__global__ void __launch_bounds__(128) mi_hist_kernel(
    const float* __restrict__ tar, const float* __restrict__ src,
    const unsigned* __restrict__ mm, float* __restrict__ partial,
    int P, int BPS) {
  __shared__ float hsh[4][1024];
  int sample = blockIdx.x / BPS;
  int blk    = blockIdx.x % BPS;
  int tid  = threadIdx.x;
  int wave = tid >> 5;
  int lane = tid & 31;
  int m    = lane & 15;
  bool laneHi = lane >= 16;

  float tmn = unflipU(mm[sample * 4 + 0]);
  float tmx = unflipU(mm[sample * 4 + 1]);
  float smn = unflipU(mm[sample * 4 + 2]);
  float smx = unflipU(mm[sample * 4 + 3]);
  float tsc = 1.0f / (tmx - tmn + 1e-12f);
  float ssc = 1.0f / (smx - smn + 1e-12f);

  float bin0 = m * BIN_STEP;          // tar-bin (A row) / src-bin (B col), tile 0
  float bin1 = (m + 16) * BIN_STEP;   // tile 1
  // quadratic-in-log2 coefficients, hoisted per lane:
  float qb0 = -2.0f * C2LOG * bin0, qc0 = C2LOG * bin0 * bin0;
  float qb1 = -2.0f * C2LOG * bin1, qc1 = C2LOG * bin1 * bin1;

  const float* t = tar + (size_t)sample * P;
  const float* s = src + (size_t)sample * P;

  v8f c00 = {}, c01 = {}, c10 = {}, c11 = {};

  int numChunks  = P >> 5;            // 32 voxels per K-step
  int wavesTotal = BPS * 4;
  for (int chunk = blk * 4 + wave; chunk < numChunks; chunk += wavesTotal) {
    int base = chunk << 5;
    float tv = (t[base + lane] - tmn) * tsc;   // lane holds voxel (base+lane)
    float sv = (s[base + lane] - smn) * ssc;
    v16h a0, a1, b0, b1;
#pragma unroll
    for (int h = 0; h < 16; ++h) {
      // A fragment, contiguous k order used here: lanes 0-15 row M=lane, K=0..15; lanes 16-31 K=16..31
      int kA = h + (laneHi ? 16 : 0);
      float tk = __shfl(tv, kA, 32);
      a0[h] = (_Float16)gaussw(tk, qb0, qc0);
      a1[h] = (_Float16)gaussw(tk, qb1, qc1);
      // B fragment, same contiguous k order as A: lanes 0-15 col N=lane, K=0..15; lanes 16-31 K=16..31
      int kB = h + (laneHi ? 16 : 0);
      float sk = __shfl(sv, kB, 32);
      b0[h] = (_Float16)gaussw(sk, qb0, qc0);
      b1[h] = (_Float16)gaussw(sk, qb1, qc1);
    }
    c00 = __builtin_amdgcn_wmma_f32_16x16x32_f16(false, a0, false, b0, (short)0, c00, false, false);
    c01 = __builtin_amdgcn_wmma_f32_16x16x32_f16(false, a0, false, b1, (short)0, c01, false, false);
    c10 = __builtin_amdgcn_wmma_f32_16x16x32_f16(false, a1, false, b0, (short)0, c10, false, false);
    c11 = __builtin_amdgcn_wmma_f32_16x16x32_f16(false, a1, false, b1, (short)0, c11, false, false);
  }

  // C/D 16x16 f32 layout: VGPR r, lanes 0-15 -> (M=r, N=lane); lanes 16-31 -> (M=r+8)
  int rowOff = laneHi ? 8 : 0;
#pragma unroll
  for (int r = 0; r < 8; ++r) {
    int M0 = r + rowOff;
    hsh[wave][(M0     ) * 32 + m     ] = c00[r];
    hsh[wave][(M0     ) * 32 + m + 16] = c01[r];
    hsh[wave][(M0 + 16) * 32 + m     ] = c10[r];
    hsh[wave][(M0 + 16) * 32 + m + 16] = c11[r];
  }
  __syncthreads();

  int rem = P & 31;                    // scalar tail (0 for 96^3)
  int tailBase = numChunks << 5;
  for (int e = tid; e < 1024; e += 128) {
    float v = hsh[0][e] + hsh[1][e] + hsh[2][e] + hsh[3][e];
    if (blk == 0 && rem) {
      int i = e >> 5, j = e & 31;
      float bi = i * BIN_STEP, bj = j * BIN_STEP;
      float qbi = -2.0f * C2LOG * bi, qci = C2LOG * bi * bi;
      float qbj = -2.0f * C2LOG * bj, qcj = C2LOG * bj * bj;
      for (int q = 0; q < rem; ++q) {
        float tq = (t[tailBase + q] - tmn) * tsc;
        float sq = (s[tailBase + q] - smn) * ssc;
        v += gaussw(tq, qbi, qci) * gaussw(sq, qbj, qcj);
      }
    }
    partial[((size_t)sample * BPS + blk) * 1024 + e] = v;
  }
}

__global__ void __launch_bounds__(256) mi_finalize_kernel(
    const float* __restrict__ partial, float* __restrict__ out,
    int BPS, int NS) {
  __shared__ float hist[1024];
  __shared__ float red[256];
  int tid = threadIdx.x;
  float acc = 0.0f;
  for (int sample = 0; sample < NS; ++sample) {
    for (int e = tid; e < 1024; e += 256) {
      float v = 0.0f;
      for (int b = 0; b < BPS; ++b)
        v += partial[((size_t)sample * BPS + b) * 1024 + e];
      hist[e] = v;
    }
    __syncthreads();
    // total mass
    float loc = 0.0f;
    for (int e = tid; e < 1024; e += 256) loc += hist[e];
    red[tid] = loc; __syncthreads();
    for (int st = 128; st > 0; st >>= 1) { if (tid < st) red[tid] += red[tid + st]; __syncthreads(); }
    float inv = 1.0f / red[0];
    __syncthreads();
    // joint entropy
    loc = 0.0f;
    for (int e = tid; e < 1024; e += 256) { float p = hist[e] * inv; loc += p * __logf(p + MI_EPS); }
    red[tid] = loc; __syncthreads();
    for (int st = 128; st > 0; st >>= 1) { if (tid < st) red[tid] += red[tid + st]; __syncthreads(); }
    float entJ = -red[0];
    __syncthreads();
    // tar marginal entropy
    float mg = 0.0f;
    if (tid < 32) {
      float pt = 0.0f;
      for (int j = 0; j < 32; ++j) pt += hist[tid * 32 + j];
      pt *= inv;
      mg = pt * __logf(pt + MI_EPS);
    }
    red[tid] = (tid < 32) ? mg : 0.0f;
    __syncthreads();
    if (tid == 0) { float e2 = 0.0f; for (int i = 0; i < 32; ++i) e2 += red[i]; red[0] = e2; }
    __syncthreads();
    float entT = -red[0];
    __syncthreads();
    // src marginal entropy
    mg = 0.0f;
    if (tid < 32) {
      float ps = 0.0f;
      for (int i = 0; i < 32; ++i) ps += hist[i * 32 + tid];
      ps *= inv;
      mg = ps * __logf(ps + MI_EPS);
    }
    red[tid] = (tid < 32) ? mg : 0.0f;
    __syncthreads();
    if (tid == 0) { float e2 = 0.0f; for (int i = 0; i < 32; ++i) e2 += red[i]; red[0] = e2; }
    __syncthreads();
    float entS = -red[0];
    __syncthreads();
    acc += (entT + entS) / entJ;
  }
  if (tid == 0) out[0] = -(acc / (float)NS);
}

extern "C" void kernel_launch(void* const* d_in, const int* in_sizes, int n_in,
                              void* d_out, int out_size, void* d_ws, size_t ws_size,
                              hipStream_t stream) {
  const float* tar = (const float*)d_in[0];
  const float* src = (const float*)d_in[1];
  const int NS = 4;                       // batch dim of (4,1,96,96,96)
  int P = in_sizes[0] / NS;               // 96^3 voxels per sample

  unsigned* mm   = (unsigned*)d_ws;                    // 16 minmax slots
  float* partial = (float*)((char*)d_ws + 256);        // per-block 32x32 partials
  size_t avail = ws_size > 256 ? ws_size - 256 : 0;
  long bps = (long)(avail / ((size_t)NS * 1024 * sizeof(float)));
  int BPS = (int)(bps < 1 ? 1 : (bps > 32 ? 32 : bps)); // blocks per sample

  mi_init_kernel<<<1, 16, 0, stream>>>(mm);
  mi_minmax_kernel<<<dim3(64, NS), 256, 0, stream>>>(tar, src, mm, P);
  mi_hist_kernel<<<NS * BPS, 128, 0, stream>>>(tar, src, mm, partial, P, BPS);
  mi_finalize_kernel<<<1, 256, 0, stream>>>(partial, (float*)d_out, BPS, NS);
}
